// NTM_6932077216215
// MI455X (gfx1250) — matrix operand order repaired and checked by reading, not compiled, not run on hardware
//
#include <hip/hip_runtime.h>

#define B_    64
#define T_    32
#define D_    64
#define N_    1024
#define V_    64
#define R_    4
#define H_    512
#define DIN_  320           // D + R*V
#define KCAT  832           // DIN + H
#define G4    2048          // 4*H
#define BLOCK 512
#define NW    16            // waves per block (wave32)
#define MB    16            // batch rows per block

typedef __bf16 bf16;
typedef __attribute__((ext_vector_type(16))) __bf16 v16bf;
typedef __attribute__((ext_vector_type(8)))  float  v8f;

// ---- bf16 weight cache layout in d_ws (element offsets) ----
#define SZ_WIH (2048*320)
#define SZ_WHH (2048*512)
#define SZ_RW  (4*70*512)
#define SZ_WW  (4*198*512)
#define SZ_OW  (64*768)
#define OFF_WIH 0
#define OFF_WHH (OFF_WIH + SZ_WIH)
#define OFF_RW  (OFF_WHH + SZ_WHH)
#define OFF_WW  (OFF_RW + SZ_RW)
#define OFF_OW  (OFF_WW + SZ_WW)
#define WTOT    (OFF_OW + SZ_OW)          // 2,301,952 elems

#define WS_MEM_OFF   ((size_t)WTOT * 2)               // 4,603,904 B (16B aligned)
#define WS_WPREV_OFF (WS_MEM_OFF + (size_t)B_*N_*V_*4) // + 16 MB

// ---- LDS arena layout (bytes) ----
#define SM_A   0                          // bf16 sA[16][832]  = 26624 B
#define SM_C   26624                      // f32  sC[16][512]  = 32768 B
#define SM_O   59392                      // f32  sO[16][288]  = 18432 B
#define SM_BIG 77824                      // f32  big[32768]   = 131072 B (gates OR cosR/cosW)
#define SMEM_BYTES (SM_BIG + 131072)      // 208896 B (< 320 KB/WGP)

__device__ __forceinline__ float sigmoidf_(float x) { return 1.f / (1.f + expf(-x)); }
__device__ __forceinline__ float softplusf_(float x) { return (x > 20.f) ? x : log1pf(expf(x)); }

__device__ __forceinline__ float waveRedSum(float v) {
#pragma unroll
  for (int o = 16; o; o >>= 1) v += __shfl_xor(v, o, 32);
  return v;
}
__device__ __forceinline__ float waveRedMax(float v) {
#pragma unroll
  for (int o = 16; o; o >>= 1) v = fmaxf(v, __shfl_xor(v, o, 32));
  return v;
}

// A-matrix 16x32 bf16 lane layout: elems e <- A[m][kc+16h+e], 16 consecutive k
__device__ __forceinline__ v16bf loadA16(const bf16* row, int kc, int h) {
  union { v16bf v; uint4 q[2]; } u;
  u.q[0] = *(const uint4*)(row + kc + 16 * h);
  u.q[1] = *(const uint4*)(row + kc + 16 * h + 8);
  return u.v;
}
// B-matrix 32x16 bf16 lane layout: elems e <- B[kc+16h+e][n], i.e. 16 consecutive k of weight row n
__device__ __forceinline__ v16bf loadB16(const bf16* p) {
  union { v16bf v; uint4 q[2]; } u;
  const uint4* q = (const uint4*)p;
  u.q[0] = q[0]; u.q[1] = q[1];
  return u.v;
}
__device__ __forceinline__ v16bf zeroB16() {
  union { v16bf v; uint4 q[2]; } u;
  u.q[0] = make_uint4(0u, 0u, 0u, 0u);
  u.q[1] = make_uint4(0u, 0u, 0u, 0u);
  return u.v;
}
__device__ __forceinline__ v8f wmma_bf16(v16bf a, v16bf b, v8f c) {
  return __builtin_amdgcn_wmma_f32_16x16x32_bf16(false, a, false, b, (short)0, c, false, false);
}

// softmax(beta*cos) -> gate with w_prev -> circular shift -> sharpen -> normalize.
// One wave owns `buf` (its batch row); cross-lane LDS reads rely on wave32 lockstep
// (all writes of a prior loop complete before later instructions issue; DS ops are in-order per wave).
__device__ __forceinline__ void addr_pass2(float* buf, const float* sOw, int pb,
                                           float* wsPrev, int lane) {
  float beta  = softplusf_(sOw[pb + 64]);
  float g     = sigmoidf_(sOw[pb + 65]);
  float s0 = sOw[pb + 66], s1 = sOw[pb + 67], s2 = sOw[pb + 68];
  float sm = fmaxf(s0, fmaxf(s1, s2));
  float e0 = expf(s0 - sm), e1 = expf(s1 - sm), e2 = expf(s2 - sm);
  float esum = e0 + e1 + e2;
  s0 = e0 / esum; s1 = e1 / esum; s2 = e2 / esum;
  float gamma = 1.f + softplusf_(sOw[pb + 69]);

  float mx = -1e30f;
  for (int ii = 0; ii < 32; ++ii) mx = fmaxf(mx, buf[ii * 32 + lane]);
  mx = waveRedMax(mx);
  float lm = beta * mx;

  float sum = 0.f;
  for (int ii = 0; ii < 32; ++ii) {
    int n = ii * 32 + lane;
    float e = expf(beta * buf[n] - lm);
    buf[n] = e; sum += e;
  }
  sum = waveRedSum(sum);
  float inv = 1.f / sum;

  for (int ii = 0; ii < 32; ++ii) {
    int n = ii * 32 + lane;
    float wc = buf[n] * inv;
    float wg = g * wc + (1.f - g) * wsPrev[n];
    buf[n] = wg;
  }

  float wp[32]; float wsum = 0.f;
#pragma unroll
  for (int ii = 0; ii < 32; ++ii) {
    int n = ii * 32 + lane;
    float wm  = buf[(n + 1023) & 1023];  // roll(+1)[n] = w_g[n-1]
    float wpl = buf[(n + 1) & 1023];     // roll(-1)[n] = w_g[n+1]
    float wss = s0 * wm + s1 * buf[n] + s2 * wpl;
    float w = powf(wss, gamma);
    wp[ii] = w; wsum += w;
  }
  wsum = waveRedSum(wsum);
  float invw = 1.f / (wsum + 1e-16f);
#pragma unroll
  for (int ii = 0; ii < 32; ++ii) {
    int n = ii * 32 + lane;
    float wf = wp[ii] * invw;
    buf[n] = wf;
    wsPrev[n] = wf;   // becomes w_prev for next timestep
  }
}

// one-shot f32 -> bf16 weight cache
__global__ void ntm_cvt(const float* __restrict__ Wih, const float* __restrict__ Whh,
                        const float* __restrict__ rW,  const float* __restrict__ wW,
                        const float* __restrict__ oW,  bf16* __restrict__ dst) {
  int stride = gridDim.x * blockDim.x;
  for (int i = blockIdx.x * blockDim.x + threadIdx.x; i < WTOT; i += stride) {
    float v;
    if      (i < OFF_WHH) v = Wih[i - OFF_WIH];
    else if (i < OFF_RW)  v = Whh[i - OFF_WHH];
    else if (i < OFF_WW)  v = rW[i - OFF_RW];
    else if (i < OFF_OW)  v = wW[i - OFF_WW];
    else                  v = oW[i - OFF_OW];
    dst[i] = (bf16)v;
  }
}

__global__ void __launch_bounds__(BLOCK)
ntm_main(const float* __restrict__ x, const float* __restrict__ mem_bias,
         const float* __restrict__ h_bias, const float* __restrict__ c_bias,
         const float* __restrict__ b_ih, const float* __restrict__ b_hh,
         const float* __restrict__ read_b, const float* __restrict__ write_b,
         const float* __restrict__ read_init, const float* __restrict__ out_b,
         const bf16* __restrict__ wbf, float* __restrict__ gmem,
         float* __restrict__ gwprev, float* __restrict__ out) {
  extern __shared__ char smem[];
  bf16*  sA  = (bf16*)(smem + SM_A);    // [16][832]: [x_t | reads(4x64) | h]
  float* sC  = (float*)(smem + SM_C);   // [16][512]
  float* sO  = (float*)(smem + SM_O);   // [16][288]: o_r cols 0..79(pad), o_w cols 80..287(pad)
  float* big = (float*)(smem + SM_BIG); // gates[16][2048]  OR  cosR[16][1024]+cosW[16][1024]

  const int tid  = threadIdx.x;
  const int wave = tid >> 5;
  const int lane = tid & 31;
  const int b0   = blockIdx.x * MB;
  const int m    = lane & 15;   // WMMA A row / B col
  const int hh   = lane >> 4;   // WMMA K half select

  const bf16* Wih = wbf + OFF_WIH;
  const bf16* Whh = wbf + OFF_WHH;
  const bf16* rWb = wbf + OFF_RW;
  const bf16* wWb = wbf + OFF_WW;
  const bf16* oWb = wbf + OFF_OW;

  // ---------------- init state (every launch; deterministic) ----------------
  {
    float4* dst = (float4*)(gmem + (size_t)b0 * (N_ * V_));
    const float4* src = (const float4*)mem_bias;
    const int tot4 = MB * N_ * V_ / 4, nv4 = N_ * V_ / 4;
    for (int idx = tid; idx < tot4; idx += BLOCK) dst[idx] = src[idx & (nv4 - 1)];
  }
  {
    float* wp = gwprev + (size_t)b0 * (2 * R_ * N_);
    for (int idx = tid; idx < MB * 2 * R_ * N_; idx += BLOCK) wp[idx] = 0.f;
  }
  for (int idx = tid; idx < MB * (R_ * V_); idx += BLOCK) {
    int mm = idx / (R_ * V_), rv = idx % (R_ * V_);
    sA[mm * KCAT + 64 + rv] = (bf16)read_init[rv];
  }
  for (int idx = tid; idx < MB * H_; idx += BLOCK) {
    int mm = idx >> 9, j = idx & 511;
    sA[mm * KCAT + DIN_ + j] = (bf16)h_bias[j];
    sC[mm * H_ + j] = c_bias[j];
  }
  __syncthreads();

  // ---------------- time loop ----------------
  for (int t = 0; t < T_; ++t) {
    // x_t -> sA[:, 0:64]
    for (int idx = tid; idx < MB * D_; idx += BLOCK) {
      int mm = idx >> 6, d = idx & 63;
      sA[mm * KCAT + d] = (bf16)x[((size_t)(b0 + mm) * T_ + t) * D_ + d];
    }
    __syncthreads();

    // ---- gates = [contr_x | h] @ [W_ih | W_hh]^T + b  (16x832x2048, bf16 WMMA) ----
    {
      const bf16* arow = sA + m * KCAT;
      for (int j = 0; j < 8; ++j) {
        int nt = wave + 16 * j;
        int nout = nt * 16 + m;
        const bf16* wr_ih = Wih + (size_t)nout * DIN_;
        const bf16* wr_hh = Whh + (size_t)nout * H_;
        v8f acc = {};
        for (int kc = 0; kc < KCAT; kc += 32) {
          v16bf a = loadA16(arow, kc, hh);
          const bf16* wp = (kc < DIN_) ? (wr_ih + kc) : (wr_hh + (kc - DIN_));
          v16bf bv = loadB16(wp + 16 * hh);
          acc = wmma_bf16(a, bv, acc);
        }
        float bias = b_ih[nout] + b_hh[nout];
#pragma unroll
        for (int r = 0; r < 8; ++r) big[(r + 8 * hh) * G4 + nout] = acc[r] + bias;
      }
    }
    __syncthreads();

    // ---- LSTM cell (f32 state) ----
    for (int idx = tid; idx < MB * H_; idx += BLOCK) {
      int mm = idx >> 9, j = idx & 511;
      const float* gr = big + mm * G4;
      float gi = gr[j], gf = gr[512 + j], gg = gr[1024 + j], go = gr[1536 + j];
      float cn = sigmoidf_(gf) * sC[mm * H_ + j] + sigmoidf_(gi) * tanhf(gg);
      float hn = sigmoidf_(go) * tanhf(cn);
      sC[mm * H_ + j] = cn;
      sA[mm * KCAT + DIN_ + j] = (bf16)hn;   // co
    }
    __syncthreads();

    // ---- heads (sequential; mem updated between heads) ----
    for (int i = 0; i < R_; ++i) {
      // (a) head projections: o_r (70 -> pad 80), o_w (198 -> pad 208), K=512, WMMA
      {
        const bf16* arow = sA + m * KCAT + DIN_;   // co
        for (int tt = wave; tt < 18; tt += NW) {
          bool isRead = tt < 5;
          int jt = isRead ? tt : tt - 5;
          int nloc = jt * 16 + m;
          int nvalid = isRead ? 70 : 198;
          bool valid = nloc < nvalid;
          const bf16* W = isRead ? (rWb + ((size_t)i * 70 + nloc) * H_)
                                 : (wWb + ((size_t)i * 198 + nloc) * H_);
          const float* bias = isRead ? (read_b + i * 70) : (write_b + i * 198);
          v8f acc = {};
          for (int kc = 0; kc < H_; kc += 32) {
            v16bf a = loadA16(arow, kc, hh);
            v16bf bv = valid ? loadB16(W + kc + 16 * hh) : zeroB16();
            acc = wmma_bf16(a, bv, acc);
          }
          int colbase = isRead ? (tt * 16) : (80 + jt * 16);
          float bv2 = valid ? bias[nloc] : 0.f;
#pragma unroll
          for (int r = 0; r < 8; ++r) sO[(r + 8 * hh) * 288 + colbase + m] = acc[r] + bv2;
        }
      }
      __syncthreads();

      // (b..e) wave == batch row; everything below is per-row private
      {
        const int gb = b0 + wave;
        float* memB = gmem + (size_t)gb * (N_ * V_);
        const float* sOw = sO + wave * 288;
        float* cosR = big + wave * 1024;
        float* cosW = big + 16384 + wave * 1024;

        // fused pass: cos(read key), cos(write key), shared ||mem_row|| in one mem sweep
        float sr = 0.f, sw = 0.f;
        for (int v = lane; v < 64; v += 32) {
          float kr = sOw[v] + 1e-16f;
          float kw = sOw[80 + v] + 1e-16f;
          sr += kr * kr; sw += kw * kw;
        }
        sr = waveRedSum(sr); sw = waveRedSum(sw);
        float nbr = fmaxf(sqrtf(sr), 1e-8f), nbw = fmaxf(sqrtf(sw), 1e-8f);
        const float4* k4r = (const float4*)(sOw);
        const float4* k4w = (const float4*)(sOw + 80);
        for (int ii = 0; ii < 32; ++ii) {
          int n = ii * 32 + lane;
          const float4* mp = (const float4*)(memB + n * 64);
          float dr = 0.f, dw = 0.f, nn = 0.f;
#pragma unroll
          for (int v4 = 0; v4 < 16; ++v4) {
            float4 mv = mp[v4], kr = k4r[v4], kw = k4w[v4];
            float a0 = mv.x + 1e-16f, a1 = mv.y + 1e-16f, a2 = mv.z + 1e-16f, a3 = mv.w + 1e-16f;
            dr += a0 * (kr.x + 1e-16f) + a1 * (kr.y + 1e-16f) + a2 * (kr.z + 1e-16f) + a3 * (kr.w + 1e-16f);
            dw += a0 * (kw.x + 1e-16f) + a1 * (kw.y + 1e-16f) + a2 * (kw.z + 1e-16f) + a3 * (kw.w + 1e-16f);
            nn += a0 * a0 + a1 * a1 + a2 * a2 + a3 * a3;
          }
          float na = fmaxf(sqrtf(nn), 1e-8f);
          cosR[n] = dr / (na * nbr);
          cosW[n] = dw / (na * nbw);
        }

        // read-head weights w, then read vector (pre-update mem)
        addr_pass2(cosR, sOw, 0, gwprev + ((size_t)gb * 8 + 2 * i + 0) * N_, lane);
        {
          int v0 = 2 * lane;
          float r0 = 0.f, r1 = 0.f;
          for (int n = 0; n < N_; ++n) {
            float wv = cosR[n];
            float2 mv = *(const float2*)(memB + n * 64 + v0);
            r0 += wv * mv.x; r1 += wv * mv.y;
          }
          sA[wave * KCAT + 64 + i * 64 + v0]     = (bf16)r0;
          sA[wave * KCAT + 64 + i * 64 + v0 + 1] = (bf16)r1;
        }

        // write-head weights w2, then erase/add memory update
        addr_pass2(cosW, sOw, 80, gwprev + ((size_t)gb * 8 + 2 * i + 1) * N_, lane);
        {
          int v0 = 2 * lane;
          float ev0 = sigmoidf_(sOw[150 + v0]), ev1 = sigmoidf_(sOw[150 + v0 + 1]);
          float av0 = sOw[214 + v0], av1 = sOw[214 + v0 + 1];
          for (int n = 0; n < N_; ++n) {
            float w2 = cosW[n];
            float2* p = (float2*)(memB + n * 64 + v0);
            float2 mv = *p;
            mv.x = mv.x * (1.f - w2 * ev0) + w2 * av0;
            mv.y = mv.y * (1.f - w2 * ev1) + w2 * av1;
            *p = mv;
          }
        }
      }
      __syncthreads();
    }

    // ---- output: y = sigmoid([co | reads] @ out_W^T + out_b), 16x768x64 WMMA ----
    if (wave < 4) {
      int nout = wave * 16 + m;
      const bf16* wrow = oWb + (size_t)nout * 768;
      const bf16* row = sA + m * KCAT;
      v8f acc = {};
      for (int kc = 0; kc < 768; kc += 32) {
        int k0 = kc + 16 * hh, k1 = kc + 16 * hh + 8;   // runs never straddle the 512 boundary
        union { v16bf v; uint4 q[2]; } u;
        u.q[0] = *(const uint4*)(row + ((k0 < 512) ? (DIN_ + k0) : (k0 - 448)));
        u.q[1] = *(const uint4*)(row + ((k1 < 512) ? (DIN_ + k1) : (k1 - 448)));
        v16bf bv = loadB16(wrow + kc + 16 * hh);
        acc = wmma_bf16(u.v, bv, acc);
      }
#pragma unroll
      for (int r = 0; r < 8; ++r) {
        int mm = r + 8 * hh;
        out[((size_t)(b0 + mm) * T_ + t) * D_ + nout] = sigmoidf_(acc[r] + out_b[nout]);
      }
    }
    __syncthreads();
  }
}

extern "C" void kernel_launch(void* const* d_in, const int* in_sizes, int n_in,
                              void* d_out, int out_size, void* d_ws, size_t ws_size,
                              hipStream_t stream) {
  (void)in_sizes; (void)n_in; (void)out_size; (void)ws_size;
  const float* x         = (const float*)d_in[0];
  const float* mem_bias  = (const float*)d_in[1];
  const float* h_bias    = (const float*)d_in[2];
  const float* c_bias    = (const float*)d_in[3];
  const float* W_ih      = (const float*)d_in[4];
  const float* W_hh      = (const float*)d_in[5];
  const float* b_ih      = (const float*)d_in[6];
  const float* b_hh      = (const float*)d_in[7];
  const float* read_W    = (const float*)d_in[8];
  const float* read_b    = (const float*)d_in[9];
  const float* write_W   = (const float*)d_in[10];
  const float* write_b   = (const float*)d_in[11];
  const float* read_init = (const float*)d_in[12];
  const float* out_W     = (const float*)d_in[13];
  const float* out_b     = (const float*)d_in[14];

  char* ws = (char*)d_ws;
  bf16*  wbf    = (bf16*)ws;
  float* gmem   = (float*)(ws + WS_MEM_OFF);
  float* gwprev = (float*)(ws + WS_WPREV_OFF);

  ntm_cvt<<<256, 256, 0, stream>>>(W_ih, W_hh, read_W, write_W, out_W, wbf);

  hipFuncSetAttribute(reinterpret_cast<const void*>(ntm_main),
                      hipFuncAttributeMaxDynamicSharedMemorySize, SMEM_BYTES);
  ntm_main<<<dim3(B_ / MB), dim3(BLOCK), SMEM_BYTES, stream>>>(
      x, mem_bias, h_bias, c_bias, b_ih, b_hh, read_b, write_b, read_init,
      out_b, wbf, gmem, gwprev, (float*)d_out);
}
